// KiloNerf_1726576854934
// MI455X (gfx1250) — matrix operand order repaired and checked by reading, not compiled, not run on hardware
//
#include <hip/hip_runtime.h>
#include <hip/hip_bf16.h>
#include <math.h>

typedef __attribute__((ext_vector_type(16))) _Float16 v16h;
typedef __attribute__((ext_vector_type(8)))  float    v8f;

#define GN 16
#define NCELL 4096
#define SCALE_F 3.0f

// ---------------- bucketing kernels ----------------

__global__ void kn_zero(int* counts) {
  counts[blockIdx.x * 256 + threadIdx.x] = 0;
}

__global__ void kn_assign(const float* __restrict__ x, int P,
                          int* __restrict__ cell, int* __restrict__ rank,
                          int* __restrict__ counts) {
  int p = blockIdx.x * 256 + threadIdx.x;
  if (p >= P) return;
  const float inv = (float)GN / SCALE_F;
  int id[3];
#pragma unroll
  for (int dd = 0; dd < 3; ++dd) {
    float t = x[p * 3 + dd] * inv + (float)(GN / 2);
    int i = (int)t; // trunc toward zero == numpy astype(int32)
    i = i < 0 ? 0 : (i > GN - 1 ? GN - 1 : i);
    id[dd] = i;
  }
  int c = (id[0] * GN + id[1]) * GN + id[2];
  cell[p] = c;
  rank[p] = atomicAdd(&counts[c], 1);
}

__global__ void kn_scan(const int* __restrict__ counts, int* __restrict__ offsets) {
  __shared__ int part[256];
  int t = threadIdx.x;
  int loc[16];
  int s = 0;
#pragma unroll
  for (int i = 0; i < 16; ++i) { loc[i] = s; s += counts[t * 16 + i]; }
  part[t] = s;
  __syncthreads();
  for (int off = 1; off < 256; off <<= 1) {
    int v = (t >= off) ? part[t - off] : 0;
    __syncthreads();
    part[t] += v;
    __syncthreads();
  }
  int excl = (t == 0) ? 0 : part[t - 1];
#pragma unroll
  for (int i = 0; i < 16; ++i) offsets[t * 16 + i] = excl + loc[i];
}

__global__ void kn_scatter(const int* __restrict__ cell, const int* __restrict__ rank,
                           const int* __restrict__ offsets, int* __restrict__ order, int P) {
  int p = blockIdx.x * 256 + threadIdx.x;
  if (p >= P) return;
  order[offsets[cell[p]] + rank[p]] = p;
}

// ---------------- CDNA5 WMMA fragment helpers (wave32) ----------------
//
// A-matrix 16x32 f16 per-lane layout:
//   lane L(0-15):  row M=L, K={0..7,16..23}; lane L+16: row M=L, K={8..15,24..31}
//   half index i = (kk&7) + ((kk&16)>>1), lane-hi bit = (kk>>3)&1
// B-matrix 32x16 f16 per-lane layout:
//   lane L(0-15): col N=L, K={0..7,16..23}; lane L+16: K={8..15,24..31} (same half order as A)
// C/D 16x16 f32: v8f elem r -> row M = r + (lane<16?0:8), col N = lane&15

// Build a B fragment in registers directly from global f32 weights (row-major K x ldN).
// Out-of-range (k >= kValid or col >= nValid) elements are zero-padded.
__device__ __forceinline__ v16h load_Bfrag(const float* __restrict__ w, int ldN,
                                           int kBase, int nBase, int kValid, int nValid,
                                           int lane) {
  int col = nBase + (lane & 15);
  int k0  = kBase + ((lane & 16) ? 8 : 0);
  v16h f;
#pragma unroll
  for (int i = 0; i < 16; ++i) {
    int k = k0 + (i & 7) + ((i & 8) << 1);
    float v = (k < kValid && col < nValid) ? w[k * ldN + col] : 0.0f;
    f[i] = (_Float16)v;
  }
  return f;
}

// Scatter one activation half into an A-fragment-ordered LDS buffer (512 halves per k-chunk).
__device__ __forceinline__ void putA(_Float16* fragHalves, int row, int k, float v) {
  int kc = k >> 5, kk = k & 31;
  int dl = ((kk >> 3) & 1) << 4;
  int ii = (kk & 7) + ((kk & 16) >> 1);
  fragHalves[kc * 512 + (row + dl) * 16 + ii] = (_Float16)v;
}

// Epilogue: bias(+relu), f32->f16, scatter C tile into next layer's A-fragment LDS layout.
__device__ __forceinline__ void store_tile_frag(v8f acc, _Float16* fragHalves, int nBase,
                                                float bias, bool relu, int lane) {
  int col = nBase + (lane & 15);   // becomes next-layer K index
  int kc  = col >> 5;
  int kk  = col & 31;
  int dl  = ((kk >> 3) & 1) << 4;
  int ii  = (kk & 7) + ((kk & 16) >> 1);
  int rbase = (lane & 16) ? 8 : 0;
  _Float16* base = fragHalves + kc * 512 + dl * 16 + ii;
#pragma unroll
  for (int r = 0; r < 8; ++r) {
    float v = acc[r] + bias;
    if (relu) v = fmaxf(v, 0.0f);
    base[(rbase + r) * 16] = (_Float16)v;
  }
}

#define WMMA(ACC, A, B) \
  (ACC) = __builtin_amdgcn_wmma_f32_16x16x32_f16(false, (A), false, (B), (short)0, (ACC), false, false)

// ---------------- main MLP kernel: one wave per grid cell ----------------

__global__ __launch_bounds__(32) void kn_mlp(
    const float* __restrict__ x, const float* __restrict__ d,
    const float* __restrict__ l1w, const float* __restrict__ l1b,
    const float* __restrict__ l2w, const float* __restrict__ l2b,
    const float* __restrict__ l3w, const float* __restrict__ l3b,
    const float* __restrict__ l4w, const float* __restrict__ l4b,
    const float* __restrict__ l5w, const float* __restrict__ l5b,
    const int* __restrict__ counts, const int* __restrict__ offsets,
    const int* __restrict__ order, float* __restrict__ out, int P) {
  // Activation buffers in A-fragment order: one v16h per lane per 32-wide k-chunk.
  __shared__ v16h fragX[2 * 32];   // pos-enc features (64 wide, 63 valid)
  __shared__ v16h fragD[2 * 32];   // chunk0: layer3 out (32), chunk1: dir-enc (27) + pad
  __shared__ v16h fragH[32];       // reused hidden activations h1/h2/h4 (32 wide)
  __shared__ float dens[16];
  __shared__ float colorL[16 * 3];
  __shared__ int rowPoint[16];
  __shared__ int rowMask[16];

  _Float16* fragXh = (_Float16*)fragX;
  _Float16* fragDh = (_Float16*)fragD;
  _Float16* fragHh = (_Float16*)fragH;

  const int c    = blockIdx.x;
  const int lane = threadIdx.x;
  const int cnt  = counts[c];
  if (cnt == 0) return;
  const int start = offsets[c];
  const int nn = lane & 15;

  // ---- per-cell weight fragments, straight global->register, each f32 read once ----
  const float* w1 = l1w + (size_t)c * 63 * 32;
  const float* w2 = l2w + (size_t)c * 32 * 33;
  const float* w3 = l3w + (size_t)c * 32 * 32;
  const float* w4 = l4w + (size_t)c * 59 * 32;
  const float* w5 = l5w + (size_t)c * 32 * 3;

  v16h B1[2][2], B2[3], B3[2], B4[2][2], B5;
#pragma unroll
  for (int n = 0; n < 2; ++n)
#pragma unroll
    for (int k = 0; k < 2; ++k)
      B1[n][k] = load_Bfrag(w1, 32, k * 32, n * 16, 63, 32, lane);
#pragma unroll
  for (int n = 0; n < 3; ++n) B2[n] = load_Bfrag(w2, 33, 0, n * 16, 32, 33, lane);
#pragma unroll
  for (int n = 0; n < 2; ++n) B3[n] = load_Bfrag(w3, 32, 0, n * 16, 32, 32, lane);
#pragma unroll
  for (int n = 0; n < 2; ++n)
#pragma unroll
    for (int k = 0; k < 2; ++k)
      B4[n][k] = load_Bfrag(w4, 32, k * 32, n * 16, 59, 32, lane);
  B5 = load_Bfrag(w5, 3, 0, 0, 32, 3, lane);

  // ---- per-lane bias values (col = nBase + nn) ----
  float bias1[2] = { l1b[(size_t)c * 32 + nn],      l1b[(size_t)c * 32 + 16 + nn] };
  float bias2[2] = { l2b[(size_t)c * 33 + nn],      l2b[(size_t)c * 33 + 16 + nn] };
  float bias2d   = (nn == 0) ? l2b[(size_t)c * 33 + 32] : 0.0f;
  float bias3[2] = { l3b[(size_t)c * 32 + nn],      l3b[(size_t)c * 32 + 16 + nn] };
  float bias4[2] = { l4b[(size_t)c * 32 + nn],      l4b[(size_t)c * 32 + 16 + nn] };
  float bias5    = (nn < 3) ? l5b[(size_t)c * 3 + nn] : 0.0f;

  for (int t0 = 0; t0 < cnt; t0 += 16) {
    // ---- stage per-point features (lanes 0..15, one row each). Tail rows clamp to
    //      the last point (row-parallel net; invalid rows simply aren't written back).
    if (lane < 16) {
      int r = lane;
      int src = t0 + r;
      int valid = src < cnt;
      int p = order[start + (valid ? src : cnt - 1)];
      rowPoint[r] = valid ? p : -1;
      float xv0 = x[p * 3 + 0], xv1 = x[p * 3 + 1], xv2 = x[p * 3 + 2];
      const float half = SCALE_F * 0.5f;
      rowMask[r] = (fabsf(xv0) < half) && (fabsf(xv1) < half) && (fabsf(xv2) < half);
      putA(fragXh, r, 0, xv0); putA(fragXh, r, 1, xv1); putA(fragXh, r, 2, xv2);
      float sc = 1.0f;
#pragma unroll
      for (int j = 0; j < 10; ++j) {
        float s0 = sc * xv0, s1 = sc * xv1, s2 = sc * xv2;
        putA(fragXh, r, 3 + 6 * j + 0, __sinf(s0));
        putA(fragXh, r, 3 + 6 * j + 1, __sinf(s1));
        putA(fragXh, r, 3 + 6 * j + 2, __sinf(s2));
        putA(fragXh, r, 3 + 6 * j + 3, __cosf(s0));
        putA(fragXh, r, 3 + 6 * j + 4, __cosf(s1));
        putA(fragXh, r, 3 + 6 * j + 5, __cosf(s2));
        sc *= 2.0f;
      }
      putA(fragXh, r, 63, 0.0f);
      // direction encoding -> fragD k-chunk 1 (k = 32..58), pad 59..63
      float dv0 = d[p * 3 + 0], dv1 = d[p * 3 + 1], dv2 = d[p * 3 + 2];
      putA(fragDh, r, 32, dv0); putA(fragDh, r, 33, dv1); putA(fragDh, r, 34, dv2);
      sc = 1.0f;
#pragma unroll
      for (int j = 0; j < 4; ++j) {
        float s0 = sc * dv0, s1 = sc * dv1, s2 = sc * dv2;
        putA(fragDh, r, 35 + 6 * j + 0, __sinf(s0));
        putA(fragDh, r, 35 + 6 * j + 1, __sinf(s1));
        putA(fragDh, r, 35 + 6 * j + 2, __sinf(s2));
        putA(fragDh, r, 35 + 6 * j + 3, __cosf(s0));
        putA(fragDh, r, 35 + 6 * j + 4, __cosf(s1));
        putA(fragDh, r, 35 + 6 * j + 5, __cosf(s2));
        sc *= 2.0f;
      }
#pragma unroll
      for (int k = 59; k < 64; ++k) putA(fragDh, r, k, 0.0f);
    }
    __syncthreads();

    // ---- layer 1: featA(16x64) @ W1, relu -> fragH ----
    {
      v16h a0 = fragX[lane], a1 = fragX[32 + lane];
#pragma unroll
      for (int n = 0; n < 2; ++n) {
        v8f acc = {};
        WMMA(acc, a0, B1[n][0]);
        WMMA(acc, a1, B1[n][1]);
        store_tile_frag(acc, fragHh, n * 16, bias1[n], true, lane);
      }
    }
    __syncthreads();

    // ---- layer 2: h1 @ W2 (32x33); cols 0..31 -> fragH (relu), col 32 -> density ----
    {
      v16h a = fragH[lane];
#pragma unroll
      for (int n = 0; n < 2; ++n) {
        v8f acc = {};
        WMMA(acc, a, B2[n]);
        store_tile_frag(acc, fragHh, n * 16, bias2[n], true, lane);
      }
      v8f accd = {};
      WMMA(accd, a, B2[2]);
      int rbase = (lane & 16) ? 8 : 0;
      if (nn == 0) {
#pragma unroll
        for (int r = 0; r < 8; ++r) dens[rbase + r] = fmaxf(accd[r] + bias2d, 0.0f);
      }
    }
    __syncthreads();

    // ---- layer 3: h2 @ W3, no relu -> fragD k-chunk 0 ----
    {
      v16h a = fragH[lane];
#pragma unroll
      for (int n = 0; n < 2; ++n) {
        v8f acc = {};
        WMMA(acc, a, B3[n]);
        store_tile_frag(acc, fragDh, n * 16, bias3[n], false, lane);
      }
    }
    __syncthreads();

    // ---- layer 4: [h3 | dir-enc](16x64) @ W4, relu -> fragH ----
    {
      v16h a0 = fragD[lane], a1 = fragD[32 + lane];
#pragma unroll
      for (int n = 0; n < 2; ++n) {
        v8f acc = {};
        WMMA(acc, a0, B4[n][0]);
        WMMA(acc, a1, B4[n][1]);
        store_tile_frag(acc, fragHh, n * 16, bias4[n], true, lane);
      }
    }
    __syncthreads();

    // ---- layer 5: h4 @ W5 (3 valid cols), sigmoid -> colorL ----
    {
      v16h a = fragH[lane];
      v8f acc = {};
      WMMA(acc, a, B5);
      int rbase = (lane & 16) ? 8 : 0;
      if (nn < 3) {
#pragma unroll
        for (int r = 0; r < 8; ++r) {
          float v = acc[r] + bias5;
          colorL[(rbase + r) * 3 + nn] = 1.0f / (1.0f + __expf(-v));
        }
      }
    }
    __syncthreads();

    // ---- masked writeback: color (P x 3) then sigma (P) ----
    if (lane < 16) {
      int r = lane;
      int p = rowPoint[r];
      if (p >= 0) {
        float m = rowMask[r] ? 1.0f : 0.0f;
#pragma unroll
        for (int j = 0; j < 3; ++j) out[p * 3 + j] = m * colorL[r * 3 + j];
        out[3 * P + p] = m * dens[r];
      }
    }
    __syncthreads();
  }
}

// ---------------- launcher ----------------

extern "C" void kernel_launch(void* const* d_in, const int* in_sizes, int n_in,
                              void* d_out, int out_size, void* d_ws, size_t ws_size,
                              hipStream_t stream) {
  (void)n_in; (void)out_size; (void)ws_size;
  const float* x   = (const float*)d_in[0];
  const float* dd  = (const float*)d_in[1];
  const float* l1w = (const float*)d_in[2];
  const float* l1b = (const float*)d_in[3];
  const float* l2w = (const float*)d_in[4];
  const float* l2b = (const float*)d_in[5];
  const float* l3w = (const float*)d_in[6];
  const float* l3b = (const float*)d_in[7];
  const float* l4w = (const float*)d_in[8];
  const float* l4b = (const float*)d_in[9];
  const float* l5w = (const float*)d_in[10];
  const float* l5b = (const float*)d_in[11];
  float* out = (float*)d_out;
  int P = in_sizes[0] / 3;

  int* counts  = (int*)d_ws;
  int* offsets = counts + NCELL;
  int* cell    = offsets + NCELL;
  int* rank    = cell + P;
  int* order   = rank + P;

  kn_zero   <<<NCELL / 256, 256, 0, stream>>>(counts);
  kn_assign <<<(P + 255) / 256, 256, 0, stream>>>(x, P, cell, rank, counts);
  kn_scan   <<<1, 256, 0, stream>>>(counts, offsets);
  kn_scatter<<<(P + 255) / 256, 256, 0, stream>>>(cell, rank, offsets, order, P);
  kn_mlp    <<<NCELL, 32, 0, stream>>>(x, dd, l1w, l1b, l2w, l2b, l3w, l3b,
                                       l4w, l4b, l5w, l5b, counts, offsets, order, out, P);
}
